// EnhancedBianGuaAttention_76055280878201
// MI455X (gfx1250) — compile-verified, matrix operand order checked by reading, not run on hardware
//
#include <hip/hip_runtime.h>
#include <hip/hip_bf16.h>

// MI455X / gfx1250, wave32. All matmuls via V_WMMA_F32_16X16X32_F16.
// GEMMs: block = 64M x 128N, W tile staged to LDS via global_load_async_to_lds
// (double buffered, ASYNCcnt) and shared by the 4 waves; A fragments from
// global. __launch_bounds__(128,1) so the 64 accumulator VGPRs + fragments
// stay resident (round-3 asm showed scratch spills without it).

typedef _Float16 v16h __attribute__((ext_vector_type(16)));
typedef _Float16 v8h  __attribute__((ext_vector_type(8)));
typedef float    v8f  __attribute__((ext_vector_type(8)));
typedef int      i4v  __attribute__((ext_vector_type(4)));

// Pointer-to-AS-qualified-int4 types for the async-copy builtin
typedef __attribute__((address_space(1))) i4v* gptr4;
typedef __attribute__((address_space(3))) i4v* lptr4;

static constexpr int Bc  = 2;
static constexpr int Tc  = 2048;
static constexpr int Dc  = 1024;
static constexpr int Hc  = 16;
static constexpr int HDc = 64;
static constexpr int ROWS = Bc * Tc;   // 4096

#define DEV __device__ __forceinline__

#if __has_builtin(__builtin_amdgcn_global_load_async_to_lds_b128)
#define HAVE_ASYNC_LDS 1
#else
#define HAVE_ASYNC_LDS 0
#endif

union H16u { v16h v; v8h h[2]; };

// Load a 16x32 f16 fragment (A-operand layout, ISA 7.12.2 "16-bit A-Matrix 16x32"):
// lane L holds row r = L&15; lanes 0-15 hold K {0..7,16..23}, lanes 16-31 hold
// K {8..15,24..31}.  Works for the B operand too (mirror layout, lane = column,
// K contiguous), i.e. any row-major [16 x K-contig] tile, global or LDS.
DEV v16h ldfrag(const _Float16* tile, int ld, int lane) {
  int r    = lane & 15;
  int koff = (lane & 16) ? 8 : 0;
  const _Float16* p = tile + (size_t)r * ld + koff;
  H16u u;
  u.h[0] = *(const v8h*)(p);
  u.h[1] = *(const v8h*)(p + 16);
  return u.v;
}

DEV v8f wmma_f16(v16h a, v16h b, v8f c) {
  return __builtin_amdgcn_wmma_f32_16x16x32_f16(
      /*neg_a=*/false, a, /*neg_b=*/false, b,
      /*c_mod=*/(short)0, c, /*reuse_a=*/false, /*reuse_b=*/false);
}

// Stage a 128(N) x 32(K) f16 weight tile (row stride Dc halves in global,
// 32 halves in LDS). 512 x 16B chunks split over 4 waves x 32 lanes x 4.
DEV void stage_tile(const _Float16* wt, _Float16* buf, int wave, int lane) {
  int base = (wave * 32 + lane) * 4;
#pragma unroll
  for (int i = 0; i < 4; i++) {
    int idx = base + i;
    int row = idx >> 2;
    int col = (idx & 3) * 8;           // in halves (16B chunks)
#if HAVE_ASYNC_LDS
    __builtin_amdgcn_global_load_async_to_lds_b128(
        (gptr4)(const void*)(wt + (size_t)row * Dc + col),
        (lptr4)(void*)(buf + row * 32 + col),
        0, 0);
#else
    *(v8h*)(buf + row * 32 + col) = *(const v8h*)(wt + (size_t)row * Dc + col);
#endif
  }
}

DEV void wait_async_4() {
#if HAVE_ASYNC_LDS
#if __has_builtin(__builtin_amdgcn_s_wait_asynccnt)
  __builtin_amdgcn_s_wait_asynccnt(4);
#else
  asm volatile("s_wait_asynccnt 0x4" ::: "memory");
#endif
#endif
}

DEV void wait_async_0() {
#if HAVE_ASYNC_LDS
#if __has_builtin(__builtin_amdgcn_s_wait_asynccnt)
  __builtin_amdgcn_s_wait_asynccnt(0);
#else
  asm volatile("s_wait_asynccnt 0x0" ::: "memory");
#endif
#endif
}

// Core: acc[8] (16M x 128N per wave) += A[m0:m0+16, :] * W[n0:n0+128, :]^T
// over K = Dc, with double-buffered LDS staging of W.
DEV void gemm_acc_16x128(const _Float16* __restrict__ A,
                         const _Float16* __restrict__ W,
                         int m0, int n0, int wave, int lane,
                         _Float16* bufs, v8f acc[8]) {
  const _Float16* a  = A + (size_t)m0 * Dc;
  const _Float16* w0 = W + (size_t)n0 * Dc;
  constexpr int KSTEPS = Dc / 32;      // 32
  stage_tile(w0, bufs, wave, lane);
  for (int s = 0; s < KSTEPS; s++) {
    int kk = s * 32;
    _Float16* cur = bufs + (size_t)(s & 1) * (128 * 32);
    if (s + 1 < KSTEPS) {
      stage_tile(w0 + kk + 32, bufs + (size_t)((s + 1) & 1) * (128 * 32), wave, lane);
      wait_async_4();                  // previous 4 async ops (current tile) done
    } else {
      wait_async_0();
    }
    __syncthreads();                   // all waves' staging visible
    __builtin_prefetch(a + kk + 256, 0, 1);   // global_prefetch_b8
    v16h af = ldfrag(a + kk, Dc, lane);
#pragma unroll
    for (int t = 0; t < 8; t++) {
      v16h bf = ldfrag(cur + t * 16 * 32, 32, lane);
      acc[t] = wmma_f16(af, bf, acc[t]);
    }
    __syncthreads();                   // done reading before buffer reuse
  }
}

// ---------------------------------------------------------------- prep ------
// scal[0]=lam, scal[1]=sigmoid(scale_logit)*2, scal[2]=alpha, scal[3]=beta
// Mh[h*36 + d*6 + e] = sum_t softmax(tw[h,:]/0.5)[t] * transforms[t,d,e]
__global__ void prep_kernel(const float* __restrict__ tw,
                            const float* __restrict__ transforms,
                            const float* __restrict__ lamlogit,
                            const float* __restrict__ scalelogit,
                            const float* __restrict__ alpha,
                            const float* __restrict__ beta,
                            float* __restrict__ scal,
                            float* __restrict__ Mh) {
  int tid = threadIdx.x;
  if (tid == 0) {
    scal[0] = 1.0f / (1.0f + __expf(-lamlogit[0]));
    scal[1] = (1.0f / (1.0f + __expf(-scalelogit[0]))) * 2.0f;
    scal[2] = alpha[0];
    scal[3] = beta[0];
  }
  if (tid < Hc * 36) {
    int h = tid / 36, de = tid % 36;
    float mx = -1e30f;
    for (int t = 0; t < 7; t++) mx = fmaxf(mx, tw[h * 7 + t] * 2.0f);
    float e[7], s = 0.0f;
    for (int t = 0; t < 7; t++) { e[t] = __expf(tw[h * 7 + t] * 2.0f - mx); s += e[t]; }
    float inv = 1.0f / s, acc = 0.0f;
    for (int t = 0; t < 7; t++) acc += (e[t] * inv) * transforms[t * 36 + de];
    Mh[h * 36 + de] = acc;
  }
}

// ---------------------------------------------------------------- convert ---
__global__ void cvt_kernel(const float* __restrict__ in, _Float16* __restrict__ out, int n) {
  int i = blockIdx.x * blockDim.x + threadIdx.x;
  if (i < n) out[i] = (_Float16)in[i];
}

// ------------------------------------------------------------- row tables ---
// Per global row r (=b*T+t): sb[r][6]=tanh(x_r.hex_w^T), hx[r][6]=tanh(x_r.q6_w^T),
// g[h][r][e] = sum_d hx[r][d]*Mh[h][d][e].   One wave per row.
__global__ void rows_kernel(const float* __restrict__ x,
                            const float* __restrict__ hex_w,
                            const float* __restrict__ q6_w,
                            const float* __restrict__ Mh,
                            float* __restrict__ sb,
                            float* __restrict__ hx,
                            float* __restrict__ g) {
  int wave = threadIdx.x >> 5, lane = threadIdx.x & 31;
  int row = blockIdx.x * 4 + wave;
  const float* xr = x + (size_t)row * Dc;
  float sbv[6], hxv[6];
#pragma unroll
  for (int j = 0; j < 6; j++) {
    const float* w1 = hex_w + j * Dc;
    const float* w2 = q6_w + j * Dc;
    float a = 0.0f, b = 0.0f;
    for (int d = lane; d < Dc; d += 32) {
      float xv = xr[d];
      a += xv * w1[d];
      b += xv * w2[d];
    }
#pragma unroll
    for (int off = 16; off; off >>= 1) {
      a += __shfl_xor(a, off, 32);
      b += __shfl_xor(b, off, 32);
    }
    sbv[j] = tanhf(a);
    hxv[j] = tanhf(b);
  }
  if (lane < 6) {
    sb[row * 6 + lane] = sbv[lane];
    hx[row * 6 + lane] = hxv[lane];
  }
  for (int idx = lane; idx < Hc * 6; idx += 32) {
    int h = idx / 6, e = idx % 6;
    float acc = 0.0f;
#pragma unroll
    for (int d = 0; d < 6; d++) acc += hxv[d] * Mh[(h * 6 + d) * 6 + e];
    g[((size_t)h * ROWS + row) * 6 + e] = acc;
  }
}

// ---------------------------------------------------------------- QKV GEMM --
// A: x_f16 [4096 x 1024], W: qkv_w_f16 [3072 x 1024] (row=n, K-contig).
// Q/K stored (b,h,t,d); V stored transposed (b,h,d,t) so the P*V WMMA can
// load K-contiguous B fragments.
__global__ void __launch_bounds__(128, 1)
gemm_qkv(const _Float16* __restrict__ A,
         const _Float16* __restrict__ W,
         _Float16* __restrict__ q,
         _Float16* __restrict__ k,
         _Float16* __restrict__ v) {
  __shared__ __align__(16) _Float16 bufs[2 * 128 * 32];
  int bm = blockIdx.x & 63;          // 64 M tiles
  int bn = blockIdx.x >> 6;          // 24 N tiles (128-wide)
  int wave = threadIdx.x >> 5, lane = threadIdx.x & 31;
  int m0 = bm * 64 + wave * 16;
  int n0 = bn * 128;
  v8f acc[8];
#pragma unroll
  for (int t = 0; t < 8; t++)
#pragma unroll
    for (int i = 0; i < 8; i++) acc[t][i] = 0.0f;

  gemm_acc_16x128(A, W, m0, n0, wave, lane, bufs, acc);

  // C layout: lane holds col n = lane&15, rows M = v + 8*(lane>=16)
  int b = m0 >> 11;
  int t0 = m0 & (Tc - 1);
  int nlane = lane & 15;
  int mhi = (lane & 16) ? 8 : 0;
  int which = n0 >> 10;              // 0=q, 1=k, 2=v (128 | 1024 so uniform)
  int nbase = n0 & (Dc - 1);
#pragma unroll
  for (int tt = 0; tt < 8; tt++) {
    int nn = nbase + tt * 16 + nlane;
    int h = nn >> 6, d = nn & 63;
#pragma unroll
    for (int vv = 0; vv < 8; vv++) {
      int trow = t0 + mhi + vv;
      _Float16 val = (_Float16)acc[tt][vv];
      if (which == 0)      q[((size_t)(b * Hc + h) * Tc + trow) * HDc + d] = val;
      else if (which == 1) k[((size_t)(b * Hc + h) * Tc + trow) * HDc + d] = val;
      else                 v[((size_t)(b * Hc + h) * HDc + d) * Tc + trow] = val;
    }
  }
}

// -------------------------------------------------------------- attention ---
// Flash-style: one wave = 16 query rows; j-step = 32 keys:
// 4 WMMAs (QK^T), pointwise bias+tanh+mask, online softmax (shfl_xor over the
// 16-lane half that shares a row), P -> LDS (f16) -> A-layout reload, 4 WMMAs (P*V).
__global__ void __launch_bounds__(128, 1)
attn_kernel(const _Float16* __restrict__ q,
            const _Float16* __restrict__ kmat,
            const _Float16* __restrict__ vmat,
            const float* __restrict__ sb,
            const float* __restrict__ hx,
            const float* __restrict__ g,
            const float* __restrict__ scal,
            _Float16* __restrict__ o) {
  __shared__ __align__(16) _Float16 pbuf[4][16 * 32];
  int wave = threadIdx.x >> 5, lane = threadIdx.x & 31;
  int nqblk = Tc / 64;
  int bh = blockIdx.x / nqblk;
  int qblk = blockIdx.x % nqblk;
  int b = bh >> 4, h = bh & 15;
  int qbase = qblk * 64 + wave * 16;

  float lam_h   = scal[0] * 0.5f;
  float s26     = scal[1] / 6.0f;
  float alpha   = scal[2];
  float invbeta = 1.0f / scal[3];

  const _Float16* qp = q + ((size_t)bh * Tc + qbase) * HDc;
  v16h qa0 = ldfrag(qp,      HDc, lane);   // d = 0..31
  v16h qa1 = ldfrag(qp + 32, HDc, lane);   // d = 32..63

  int mhi = (lane & 16) ? 8 : 0;
  int nlane = lane & 15;

  float rsb[8][6], rg[8][6];
#pragma unroll
  for (int vv = 0; vv < 8; vv++) {
    int r = b * Tc + qbase + mhi + vv;
#pragma unroll
    for (int e = 0; e < 6; e++) {
      rsb[vv][e] = sb[(size_t)r * 6 + e];
      rg[vv][e]  = g[((size_t)h * ROWS + r) * 6 + e];
    }
  }

  float mrow[8], lrow[8];
  v8f acc[4];
#pragma unroll
  for (int vv = 0; vv < 8; vv++) { mrow[vv] = -1e30f; lrow[vv] = 0.0f; }
#pragma unroll
  for (int t = 0; t < 4; t++)
#pragma unroll
    for (int i = 0; i < 8; i++) acc[t][i] = 0.0f;

  const _Float16* kb = kmat + (size_t)bh * Tc * HDc;   // (t, d)
  const _Float16* vb = vmat + (size_t)bh * HDc * Tc;   // (d, t)
  _Float16* pb = pbuf[wave];

  for (int jb = 0; jb < qbase + 16; jb += 32) {
    v8f s[2];
#pragma unroll
    for (int sub = 0; sub < 2; sub++) {
#pragma unroll
      for (int i = 0; i < 8; i++) s[sub][i] = 0.0f;
      const _Float16* kt = kb + (size_t)(jb + sub * 16) * HDc;
      v16h k0 = ldfrag(kt,      HDc, lane);
      v16h k1 = ldfrag(kt + 32, HDc, lane);
      s[sub] = wmma_f16(qa0, k0, s[sub]);
      s[sub] = wmma_f16(qa1, k1, s[sub]);
    }
    // pointwise: alpha*tanh(raw/beta) + lam*ham/2 + scale2*lat/6, causal mask
#pragma unroll
    for (int sub = 0; sub < 2; sub++) {
      int jcol = jb + sub * 16 + nlane;
      int jr = b * Tc + jcol;
      float csb[6], chx[6];
#pragma unroll
      for (int e = 0; e < 6; e++) {
        csb[e] = sb[(size_t)jr * 6 + e];
        chx[e] = hx[(size_t)jr * 6 + e];
      }
#pragma unroll
      for (int vv = 0; vv < 8; vv++) {
        float val = alpha * tanhf(s[sub][vv] * invbeta);
        float bias = 0.0f;
#pragma unroll
        for (int e = 0; e < 6; e++)
          bias += lam_h * rsb[vv][e] * csb[e] + s26 * rg[vv][e] * chx[e];
        val += bias;
        int irow = qbase + mhi + vv;
        if (jcol > irow) val = -1e30f;
        s[sub][vv] = val;
      }
    }
    // online softmax; each row lives in one VGPR across the 16-lane half
#pragma unroll
    for (int vv = 0; vv < 8; vv++) {
      float tmax = fmaxf(s[0][vv], s[1][vv]);
#pragma unroll
      for (int off = 8; off; off >>= 1) tmax = fmaxf(tmax, __shfl_xor(tmax, off, 32));
      float mnew = fmaxf(mrow[vv], tmax);
      float corr = __expf(mrow[vv] - mnew);
      float p0 = __expf(s[0][vv] - mnew);
      float p1 = __expf(s[1][vv] - mnew);
      float psum = p0 + p1;
#pragma unroll
      for (int off = 8; off; off >>= 1) psum += __shfl_xor(psum, off, 32);
      lrow[vv] = lrow[vv] * corr + psum;
      mrow[vv] = mnew;
#pragma unroll
      for (int t = 0; t < 4; t++) acc[t][vv] *= corr;
      s[0][vv] = p0;
      s[1][vv] = p1;
    }
    // relayout P: C-layout f32 -> LDS 16x32 f16 -> A-layout fragment
#pragma unroll
    for (int sub = 0; sub < 2; sub++)
#pragma unroll
      for (int vv = 0; vv < 8; vv++)
        pb[(mhi + vv) * 32 + sub * 16 + nlane] = (_Float16)s[sub][vv];
    asm volatile("s_wait_dscnt 0" ::: "memory");
    v16h pf = ldfrag(pb, 32, lane);
    // P(16x32 over j) x V(32 j x 64 d): V rows are d with t contiguous
#pragma unroll
    for (int t = 0; t < 4; t++) {
      v16h vf = ldfrag(vb + (size_t)(t * 16) * Tc + jb, Tc, lane);
      acc[t] = wmma_f16(pf, vf, acc[t]);
    }
  }
  // normalize and write O as f16 (b, t, h*64+d) for the out-projection GEMM
#pragma unroll
  for (int vv = 0; vv < 8; vv++) {
    float inv = 1.0f / lrow[vv];
    int trow = qbase + mhi + vv;
#pragma unroll
    for (int t = 0; t < 4; t++)
      o[((size_t)(b * Tc + trow)) * Dc + h * HDc + t * 16 + nlane] =
          (_Float16)(acc[t][vv] * inv);
  }
}

// ---------------------------------------------------------------- out GEMM --
__global__ void __launch_bounds__(128, 1)
gemm_out(const _Float16* __restrict__ A,
         const _Float16* __restrict__ W,
         float* __restrict__ out) {
  __shared__ __align__(16) _Float16 bufs[2 * 128 * 32];
  int bm = blockIdx.x & 63;          // 64 M tiles
  int bn = blockIdx.x >> 6;          // 8 N tiles (128-wide)
  int wave = threadIdx.x >> 5, lane = threadIdx.x & 31;
  int m0 = bm * 64 + wave * 16;
  int n0 = bn * 128;
  v8f acc[8];
#pragma unroll
  for (int t = 0; t < 8; t++)
#pragma unroll
    for (int i = 0; i < 8; i++) acc[t][i] = 0.0f;

  gemm_acc_16x128(A, W, m0, n0, wave, lane, bufs, acc);

  int nlane = lane & 15;
  int mhi = (lane & 16) ? 8 : 0;
#pragma unroll
  for (int tt = 0; tt < 8; tt++) {
    int n = n0 + tt * 16 + nlane;
#pragma unroll
    for (int vv = 0; vv < 8; vv++) {
      int m = m0 + mhi + vv;
      out[(size_t)m * Dc + n] = acc[tt][vv];
    }
  }
}

// ---------------------------------------------------------------- launch ----
extern "C" void kernel_launch(void* const* d_in, const int* in_sizes, int n_in,
                              void* d_out, int out_size, void* d_ws, size_t ws_size,
                              hipStream_t stream) {
  const float* x          = (const float*)d_in[0];
  const float* qkv_w      = (const float*)d_in[1];
  const float* out_w      = (const float*)d_in[2];
  const float* hex_w      = (const float*)d_in[3];
  const float* lamlogit   = (const float*)d_in[4];
  const float* q6_w       = (const float*)d_in[5];
  const float* transforms = (const float*)d_in[6];
  const float* tw         = (const float*)d_in[7];
  const float* scalelogit = (const float*)d_in[8];
  const float* alpha      = (const float*)d_in[9];
  const float* beta       = (const float*)d_in[10];
  float* out = (float*)d_out;
  (void)in_sizes; (void)n_in; (void)out_size; (void)ws_size;

  char* ws = (char*)d_ws;
  auto al = [](size_t v) { return (v + 255) & ~size_t(255); };
  size_t off = 0;
  float*    scal = (float*)(ws + off); off = al(off + 4 * sizeof(float));
  float*    Mh   = (float*)(ws + off); off = al(off + Hc * 36 * sizeof(float));
  float*    sb   = (float*)(ws + off); off = al(off + (size_t)ROWS * 6 * sizeof(float));
  float*    hx   = (float*)(ws + off); off = al(off + (size_t)ROWS * 6 * sizeof(float));
  float*    g    = (float*)(ws + off); off = al(off + (size_t)Hc * ROWS * 6 * sizeof(float));
  _Float16* xh   = (_Float16*)(ws + off); off = al(off + (size_t)ROWS * Dc * 2);
  _Float16* wqk  = (_Float16*)(ws + off); off = al(off + (size_t)3 * Dc * Dc * 2);
  _Float16* wo   = (_Float16*)(ws + off); off = al(off + (size_t)Dc * Dc * 2);
  _Float16* qh   = (_Float16*)(ws + off); off = al(off + (size_t)ROWS * Dc * 2);
  _Float16* kh   = (_Float16*)(ws + off); off = al(off + (size_t)ROWS * Dc * 2);
  _Float16* vh   = (_Float16*)(ws + off); off = al(off + (size_t)ROWS * Dc * 2);
  _Float16* oh   = (_Float16*)(ws + off); off = al(off + (size_t)ROWS * Dc * 2);

  prep_kernel<<<1, 640, 0, stream>>>(tw, transforms, lamlogit, scalelogit, alpha, beta, scal, Mh);
  cvt_kernel<<<(ROWS * Dc) / 256, 256, 0, stream>>>(x, xh, ROWS * Dc);
  cvt_kernel<<<(3 * Dc * Dc) / 256, 256, 0, stream>>>(qkv_w, wqk, 3 * Dc * Dc);
  cvt_kernel<<<(Dc * Dc) / 256, 256, 0, stream>>>(out_w, wo, Dc * Dc);
  rows_kernel<<<ROWS / 4, 128, 0, stream>>>(x, hex_w, q6_w, Mh, sb, hx, g);
  gemm_qkv<<<(ROWS / 64) * (3 * Dc / 128), 128, 0, stream>>>(xh, wqk, qh, kh, vh);
  attn_kernel<<<Bc * Hc * (Tc / 64), 128, 0, stream>>>(qh, kh, vh, sb, hx, g, scal, oh);
  gemm_out<<<(ROWS / 64) * (Dc / 128), 128, 0, stream>>>(oh, wo, out);
}
